// PNA_58660663329439
// MI455X (gfx1250) — matrix operand order repaired and checked by reading, not compiled, not run on hardware
//
#include <hip/hip_runtime.h>
#include <stdint.h>

#define H 128

typedef __attribute__((ext_vector_type(16))) __bf16 v16bf;
typedef __attribute__((ext_vector_type(8)))  __bf16 v8bf;
typedef __attribute__((ext_vector_type(8)))  float  v8f;

#define NEG_ENC 0x007FFFFFu  // fenc(-inf)

__device__ __forceinline__ unsigned short f2bf(float f) {
    union { float f; unsigned int u; } v; v.f = f;
    unsigned int r = v.u + 0x7fffu + ((v.u >> 16) & 1u);
    return (unsigned short)(r >> 16);
}
__device__ __forceinline__ unsigned int fenc(float f) {
    unsigned int u = __float_as_uint(f);
    return (u & 0x80000000u) ? ~u : (u | 0x80000000u);
}
__device__ __forceinline__ float fdec(unsigned int u) {
    u = (u & 0x80000000u) ? (u & 0x7FFFFFFFu) : ~u;
    return __uint_as_float(u);
}

// ---------------- utility kernels ----------------
__global__ void k_fill_f32(float* p, float v, int n) {
    int i = blockIdx.x * blockDim.x + threadIdx.x;
    if (i < n) p[i] = v;
}
__global__ void k_fill_u32(unsigned int* p, unsigned int v, int n) {
    int i = blockIdx.x * blockDim.x + threadIdx.x;
    if (i < n) p[i] = v;
}
__global__ void k_deg(const int* __restrict__ col, float* __restrict__ deg, int E) {
    int e = blockIdx.x * blockDim.x + threadIdx.x;
    if (e < E) atomicAdd(&deg[col[e]], 1.0f);
}
__global__ void k_ampatt(const float* __restrict__ deg, float* __restrict__ amp,
                         float* __restrict__ att, int n) {
    int i = blockIdx.x * blockDim.x + threadIdx.x;
    if (i < n) {
        float ld = logf(deg[i] + 1.0f);
        amp[i] = ld;                      // AVG_D_LOG == 1.0
        att[i] = 1.0f / fmaxf(ld, 1e-8f);
    }
}
__global__ void k_embed(const int* __restrict__ x, const float* __restrict__ emb,
                        float* __restrict__ h, int N) {
    int idx = blockIdx.x * blockDim.x + threadIdx.x;
    if (idx < N * H) {
        int i = idx / H, c = idx % H;
        h[idx] = emb[x[i] * H + c];
    }
}
// Pre-permute fp32 weights (Ktot x 128 per layer) into per-lane WMMA bf16
// fragment order: [kb][ntile][lane][j] where K = kb*32 + (lane>>4)*16 + j,
// n = ntile*16 + (lane&15). Each wave then reads its B fragment contiguously.
__global__ void k_permW(const float* __restrict__ src, unsigned short* __restrict__ dst,
                        int Ktot, int total) {
    int idx = blockIdx.x * blockDim.x + threadIdx.x;
    if (idx >= total) return;
    int perL = Ktot * H;
    int l = idx / perL;
    int rem = idx % perL;
    int blk = rem / 512;
    int within = rem % 512;
    int kb = blk / 8, nt = blk % 8;
    int lane = within / 16, j = within % 16;
    int K = kb * 32 + (lane >> 4) * 16 + j;
    int n = nt * 16 + (lane & 15);
    dst[idx] = f2bf(src[(size_t)(l * Ktot + K) * H + n]);
}

__device__ __forceinline__ v16bf ldsAfrag(const unsigned short* sArow, int kb, int kh) {
    return *(const v16bf*)&sArow[kb * 32 + kh * 16];
}

// ---------------- message GEMM + scatter (WMMA bf16) ----------------
// One block = one 64-edge tile (4 x 16-row M sub-tiles). 8 waves x 16 cols.
// K = 384 (12 steps); per K step each wave loads one B fragment and all 4
// A fragments, then issues 4 independent WMMAs (48 WMMAs / block / wave).
#define MSG_MT 4
__global__ __launch_bounds__(256)
void k_msg(const float* __restrict__ h, const int* __restrict__ row,
           const int* __restrict__ col, const int* __restrict__ eattr,
           const float* __restrict__ bondEmb,
           const unsigned short* __restrict__ Wp, const float* __restrict__ bias,
           float* __restrict__ aggSum, unsigned int* __restrict__ aggMax, int E)
{
    extern __shared__ unsigned short sA[];          // [64][384] bf16 = 48 KB
    __shared__ int sRow[16 * MSG_MT], sCol[16 * MSG_MT], sEA[16 * MSG_MT];
    const int tid = threadIdx.x;
    const int lane = tid & 31;
    const int wv = tid >> 5;
    const int tileBase = blockIdx.x * (16 * MSG_MT);

    if (tid < 16 * MSG_MT) {
        int eM = tileBase + tid;
        sRow[tid] = (eM < E) ? row[eM]   : 0;
        sCol[tid] = (eM < E) ? col[eM]   : 0;
        sEA[tid]  = (eM < E) ? eattr[eM] : 0;
    }
    __syncthreads();

    // stage A tile: [h[row] | h[col] | e] as bf16
    for (int idx = tid; idx < 16 * MSG_MT * 384; idx += 256) {
        int m = idx / 384, k = idx % 384;
        float val;
        if (k < 128)      val = h[(size_t)sRow[m] * H + k];
        else if (k < 256) val = h[(size_t)sCol[m] * H + (k - 128)];
        else              val = bondEmb[sEA[m] * H + (k - 256)];
        sA[m * 384 + k] = f2bf(val);
    }
    __syncthreads();

    const int m  = lane & 15;
    const int kh = lane >> 4;
    v8f acc[MSG_MT];
    #pragma unroll
    for (int mt = 0; mt < MSG_MT; ++mt) acc[mt] = (v8f){};

    for (int kb = 0; kb < 12; ++kb) {
        v16bf b = *(const v16bf*)&Wp[((size_t)(kb * 8 + wv) * 32 + lane) * 16];
        v16bf a[MSG_MT];
        #pragma unroll
        for (int mt = 0; mt < MSG_MT; ++mt)
            a[mt] = ldsAfrag(&sA[(mt * 16 + m) * 384], kb, kh);
        #pragma unroll
        for (int mt = 0; mt < MSG_MT; ++mt)
            acc[mt] = __builtin_amdgcn_wmma_f32_16x16x32_bf16(false, a[mt], false, b,
                                                              (short)0, acc[mt],
                                                              false, false);
    }

    const int n  = lane & 15;
    const int cb = wv * 16 + n;
    const float bv = bias[cb];
    #pragma unroll
    for (int mt = 0; mt < MSG_MT; ++mt) {
        #pragma unroll
        for (int r = 0; r < 8; ++r) {
            int em = mt * 16 + r + 8 * kh;        // edge row in tile (M index)
            int eM = tileBase + em;
            if (eM < E) {
                float val = acc[mt][r] + bv;
                int c = sCol[em];
                atomicAdd(&aggSum[(size_t)c * H + cb], val);
                atomicMax(&aggMax[(size_t)c * H + cb], fenc(val));
            }
        }
    }
}

// ---------------- post GEMM (WMMA bf16) ----------------
// One block = one 32-node tile (2 x 16 M sub-tiles). K = 1280 (40 steps),
// B fragment reused across both sub-tiles. hOut = GEMM + b + h_in.
#define POST_MT 2
__global__ __launch_bounds__(256)
void k_post(const float* __restrict__ hIn, const float* __restrict__ aggSum,
            const unsigned int* __restrict__ aggMax, const float* __restrict__ deg,
            const float* __restrict__ amp, const float* __restrict__ att,
            const unsigned short* __restrict__ Wp, const float* __restrict__ bias,
            float* __restrict__ hOut, int N)
{
    extern __shared__ unsigned short sA[];          // [32][1280] bf16 = 80 KB
    const int tid = threadIdx.x;
    const int lane = tid & 31;
    const int wv = tid >> 5;
    const int tileBase = blockIdx.x * (16 * POST_MT);

    for (int idx = tid; idx < 16 * POST_MT * 1280; idx += 256) {
        int m = idx / 1280, k = idx % 1280;
        int node = tileBase + m;
        float val = 0.0f;
        if (node < N) {
            if (k < 1152) {
                int seg3 = k / 384;          // 0: hagg, 1: hagg*amp, 2: hagg*att
                int kk = k % 384;
                int sub = kk / 128, c = kk % 128;
                float d = deg[node];
                float s = aggSum[(size_t)node * H + c];
                float hv;
                if (sub == 0)      hv = s / fmaxf(d, 1.0f);
                else if (sub == 1) hv = s;
                else               hv = (d > 0.0f) ? fdec(aggMax[(size_t)node * H + c]) : 0.0f;
                float fac = (seg3 == 0) ? 1.0f : ((seg3 == 1) ? amp[node] : att[node]);
                val = hv * fac;
            } else {
                val = hIn[(size_t)node * H + (k - 1152)];
            }
        }
        sA[m * 1280 + k] = f2bf(val);
    }
    __syncthreads();

    const int m  = lane & 15;
    const int kh = lane >> 4;
    v8f acc[POST_MT];
    #pragma unroll
    for (int mt = 0; mt < POST_MT; ++mt) acc[mt] = (v8f){};

    for (int kb = 0; kb < 40; ++kb) {
        v16bf b = *(const v16bf*)&Wp[((size_t)(kb * 8 + wv) * 32 + lane) * 16];
        v16bf a[POST_MT];
        #pragma unroll
        for (int mt = 0; mt < POST_MT; ++mt)
            a[mt] = ldsAfrag(&sA[(mt * 16 + m) * 1280], kb, kh);
        #pragma unroll
        for (int mt = 0; mt < POST_MT; ++mt)
            acc[mt] = __builtin_amdgcn_wmma_f32_16x16x32_bf16(false, a[mt], false, b,
                                                              (short)0, acc[mt],
                                                              false, false);
    }

    const int n  = lane & 15;
    const int cb = wv * 16 + n;
    const float bv = bias[cb];
    #pragma unroll
    for (int mt = 0; mt < POST_MT; ++mt) {
        #pragma unroll
        for (int r = 0; r < 8; ++r) {
            int node = tileBase + mt * 16 + (r + 8 * kh);
            if (node < N)
                hOut[(size_t)node * H + cb] = acc[mt][r] + bv + hIn[(size_t)node * H + cb];
        }
    }
}

// ---------------- graph pooling ----------------
__global__ void k_pool(const float* __restrict__ h, const int* __restrict__ batch,
                       float* __restrict__ pSum, unsigned int* __restrict__ pMax,
                       float* __restrict__ pCnt, int N)
{
    int idx = blockIdx.x * blockDim.x + threadIdx.x;
    if (idx >= N * H) return;
    int i = idx / H, c = idx % H;
    int g = batch[i];
    float v = h[idx];
    atomicAdd(&pSum[g * H + c], v);
    atomicMax(&pMax[g * H + c], fenc(v));
    if (c == 0) atomicAdd(&pCnt[g], 1.0f);
}

// ---------------- output head (tiny: G=64) ----------------
__global__ __launch_bounds__(128)
void k_head(const float* __restrict__ pSum, const unsigned int* __restrict__ pMax,
            const float* __restrict__ pCnt, const float* __restrict__ W1,
            const float* __restrict__ b1, const float* __restrict__ W2,
            const float* __restrict__ b2, float* __restrict__ out)
{
    __shared__ float sR[384];
    __shared__ float sT[128];
    int g = blockIdx.x;
    int j = threadIdx.x;
    float cnt = pCnt[g];
    for (int k = j; k < 384; k += 128) {
        int sub = k / 128, c = k % 128;
        float s = pSum[g * H + c];
        float v;
        if (sub == 0)      v = s / fmaxf(cnt, 1.0f);
        else if (sub == 1) v = s;
        else               v = (cnt > 0.0f) ? fdec(pMax[g * H + c]) : 0.0f;
        if (isnan(v) || isinf(v)) v = 0.0f;
        sR[k] = v;
    }
    __syncthreads();
    float accj = b1[j];
    for (int k = 0; k < 384; ++k) accj += sR[k] * W1[k * H + j];
    accj = fmaxf(accj, 0.0f);
    sT[j] = accj * W2[j];
    __syncthreads();
    for (int s = 64; s > 0; s >>= 1) {
        if (j < s) sT[j] += sT[j + s];
        __syncthreads();
    }
    if (j == 0) {
        float o = sT[0] + b2[0];
        if (isnan(o) || isinf(o)) o = 0.0f;
        out[g] = o;
    }
}

extern "C" void kernel_launch(void* const* d_in, const int* in_sizes, int n_in,
                              void* d_out, int out_size, void* d_ws, size_t ws_size,
                              hipStream_t stream)
{
    const int*   x       = (const int*)d_in[0];
    const int*   eidx    = (const int*)d_in[1];
    const int*   eattr   = (const int*)d_in[2];
    const int*   batch   = (const int*)d_in[3];
    const float* atomEmb = (const float*)d_in[4];
    const float* bondEmb = (const float*)d_in[5];
    const float* preW    = (const float*)d_in[6];
    const float* preB    = (const float*)d_in[7];
    const float* postW   = (const float*)d_in[8];
    const float* postB   = (const float*)d_in[9];
    const float* outW1   = (const float*)d_in[10];
    const float* outB1   = (const float*)d_in[11];
    const float* outW2   = (const float*)d_in[12];
    const float* outB2   = (const float*)d_in[13];
    float* out = (float*)d_out;

    const int N = in_sizes[0];
    const int E = in_sizes[2];
    const int L = in_sizes[7] / H;
    const int G = out_size;
    const int* row = eidx;
    const int* col = eidx + E;

    // workspace carve (256B aligned)
    char* wp = (char*)d_ws;
    auto carve = [&](size_t bytes) -> void* {
        void* r = (void*)wp;
        wp += (bytes + 255) & ~(size_t)255;
        return r;
    };
    float*          hA      = (float*)carve((size_t)N * H * 4);
    float*          hB      = (float*)carve((size_t)N * H * 4);
    float*          aggSum  = (float*)carve((size_t)N * H * 4);
    unsigned int*   aggMax  = (unsigned int*)carve((size_t)N * H * 4);
    float*          deg     = (float*)carve((size_t)N * 4);
    float*          amp     = (float*)carve((size_t)N * 4);
    float*          att     = (float*)carve((size_t)N * 4);
    unsigned short* preWp   = (unsigned short*)carve((size_t)L * 384 * H * 2);
    unsigned short* postWp  = (unsigned short*)carve((size_t)L * 1280 * H * 2);
    float*          pSum    = (float*)carve((size_t)G * H * 4);
    unsigned int*   pMax    = (unsigned int*)carve((size_t)G * H * 4);
    float*          pCnt    = (float*)carve((size_t)G * 4);
    (void)ws_size; (void)n_in;

    // degree + scale factors
    k_fill_f32<<<(N + 255) / 256, 256, 0, stream>>>(deg, 0.f, N);
    k_deg<<<(E + 255) / 256, 256, 0, stream>>>(col, deg, E);
    k_ampatt<<<(N + 255) / 256, 256, 0, stream>>>(deg, amp, att, N);
    // node embeddings
    k_embed<<<(N * H + 255) / 256, 256, 0, stream>>>(x, atomEmb, hA, N);
    // weight prep: fp32 -> bf16, permuted into WMMA fragment order
    k_permW<<<(L * 384 * H + 255) / 256, 256, 0, stream>>>(preW, preWp, 384, L * 384 * H);
    k_permW<<<(L * 1280 * H + 255) / 256, 256, 0, stream>>>(postW, postWp, 1280, L * 1280 * H);

    float* hCur = hA; float* hNext = hB;
    const int nTilesE = (E + (16 * MSG_MT) - 1) / (16 * MSG_MT);
    const int nTilesN = (N + (16 * POST_MT) - 1) / (16 * POST_MT);
    const size_t msgLds  = (size_t)16 * MSG_MT * 384 * sizeof(unsigned short);   // 48 KB
    const size_t postLds = (size_t)16 * POST_MT * 1280 * sizeof(unsigned short); // 80 KB
    for (int l = 0; l < L; ++l) {
        k_fill_f32<<<(N * H + 255) / 256, 256, 0, stream>>>(aggSum, 0.f, N * H);
        k_fill_u32<<<(N * H + 255) / 256, 256, 0, stream>>>(aggMax, NEG_ENC, N * H);
        k_msg<<<nTilesE, 256, msgLds, stream>>>(hCur, row, col, eattr, bondEmb,
                                                preWp + (size_t)l * 384 * H, preB + l * H,
                                                aggSum, aggMax, E);
        k_post<<<nTilesN, 256, postLds, stream>>>(hCur, aggSum, aggMax, deg, amp, att,
                                                  postWp + (size_t)l * 1280 * H, postB + l * H,
                                                  hNext, N);
        float* t = hCur; hCur = hNext; hNext = t;
    }

    k_fill_f32<<<(G * H + 255) / 256, 256, 0, stream>>>(pSum, 0.f, G * H);
    k_fill_u32<<<(G * H + 255) / 256, 256, 0, stream>>>(pMax, NEG_ENC, G * H);
    k_fill_f32<<<(G + 255) / 256, 256, 0, stream>>>(pCnt, 0.f, G);
    k_pool<<<(N * H + 255) / 256, 256, 0, stream>>>(hCur, batch, pSum, pMax, pCnt, N);
    k_head<<<G, 128, 0, stream>>>(pSum, pMax, pCnt, outW1, outB1, outW2, outB2, out);
}
